// AudioEncoder_9474697855668
// MI455X (gfx1250) — matrix operand order repaired and checked by reading, not compiled, not run on hardware
//
#include <hip/hip_runtime.h>
#include <hip/hip_bf16.h>

// ---------------- problem constants ----------------
#define BATCH 2
#define MELS 128
#define CTX 1500
#define DIMS 1024
#define HEADS 16
#define HD 64
#define LAYERS 4
#define NSAMP 150000
#define FF (4*DIMS)

#define USE_TDM 1

// ---------------- WMMA types ----------------
typedef __bf16 v16bf __attribute__((ext_vector_type(16)));
typedef __bf16 v8bf  __attribute__((ext_vector_type(8)));
typedef float  v8f   __attribute__((ext_vector_type(8)));
typedef unsigned int u32x4 __attribute__((ext_vector_type(4)));
typedef int          i32x8 __attribute__((ext_vector_type(8)));

__device__ __forceinline__ unsigned short f2bf(float f) {
  unsigned int u = __builtin_bit_cast(unsigned int, f);
  unsigned int r = (u + 0x7fffu + ((u >> 16) & 1u)) >> 16;
  return (unsigned short)r;
}
__device__ __forceinline__ unsigned int pack2bf(float a, float b) {
  return (unsigned int)f2bf(a) | ((unsigned int)f2bf(b) << 16);
}
__device__ __forceinline__ float gelu_f(float x) {
  float x3 = x * x * x;
  return 0.5f * x * (1.f + tanhf(0.7978845608028654f * (x + 0.044715f * x3)));
}

// force a (workgroup-uniform) value into the SGPR register class
__device__ __forceinline__ int rfl(unsigned v) {
  return __builtin_amdgcn_readfirstlane((int)v);
}

// ---- Tensor Data Mover: issue one async 2D fp32 tile -> LDS ----
// Descriptor per CDNA5 ISA ch.8: group0 = {count, lds_addr, global_addr, type},
// group1 = {data_size, tensor_dim0/1 (OOB zero-fill), tile_dim0/1, dim0 stride}.
// Every word goes through readfirstlane so the 4/8-wide tuples can only be
// allocated as SGPR groups (tensor_load_to_lds takes scalar operands).
__device__ __forceinline__ void tdm_issue(const float* base, const void* ldsPtr,
                                          int remCols, int remRows,
                                          long strideElems, int tileCols, int tileRows)
{
  unsigned long long ga = (unsigned long long)(const void*)base;
  unsigned ldsa = (unsigned)(unsigned long long)ldsPtr;   // low 32 bits = LDS offset
  unsigned td0 = (unsigned)remCols;
  unsigned td1 = (unsigned)remRows;
  unsigned long long str = (unsigned long long)strideElems;
  u32x4 g0;
  g0[0] = (unsigned)rfl(1u);                              // count=1, user descriptor
  g0[1] = (unsigned)rfl(ldsa);                            // lds_addr
  g0[2] = (unsigned)rfl((unsigned)(ga & 0xFFFFFFFFu));    // global_addr lo
  g0[3] = (unsigned)rfl((unsigned)((ga >> 32) & 0x1FFFFFFu) | (2u << 30)); // hi | type=2
  i32x8 g1;
  g1[0] = rfl(2u << 16);                                  // data_size = 4B
  g1[1] = rfl((td0 & 0xFFFFu) << 16);                     // tensor_dim0 lo
  g1[2] = rfl((td0 >> 16) | ((td1 & 0xFFFFu) << 16));     // dim0 hi | dim1 lo
  g1[3] = rfl((td1 >> 16) | ((unsigned)tileCols << 16));  // dim1 hi | tile_dim0
  g1[4] = rfl((unsigned)tileRows);                        // tile_dim1
  g1[5] = rfl((unsigned)(str & 0xFFFFFFFFu));             // dim0 stride lo
  g1[6] = rfl((unsigned)((str >> 32) & 0xFFFFu));         // stride hi
  g1[7] = rfl(0u);
  asm volatile("tensor_load_to_lds %0, %1" : : "s"(g0), "s"(g1) : "memory");
}

// ---------------- generic batched WMMA GEMM ----------------
// C[m,n] = act( sum_k A(m,k) * B(n,k or k,n) + bias[n] ) + residual
// AMODE  : 0 = dense A[m*lda+k] (TDM-fed), 1 = implicit im2col over [C,Lin]
// TRANSB : 1 = B is [N,K] (weight, y = x W^T), 0 = B is [K,N]
// CSTORT : 1 = store C[n*ldc+m] ([B,C,L] conv layout), 0 = C[m*ldc+n]
// ACT    : 0 none, 1 relu, 2 gelu
// KSZ    : compile-time conv kernel size (AMODE==1), 0 otherwise
// TDM    : 1 = operand panels staged by tensor_load_to_lds (async tensor DMA)
struct GemmP {
  const float* A; const float* B; const float* bias; const float* res; float* C;
  int M, N, K;
  int lda, ldb, ldc;
  int innerCount;
  long aIn, aOut, bIn, bOut, cIn, cOut;
  int convLin, convStride, convPad, convK, convDil;
};

#define TILE_M 128
#define TILE_N 64
#define TILE_K 32

template<int KSZ>
__device__ __forceinline__ float fetchConv(const float* Ab, int mg, int kg, const GemmP& p) {
  constexpr int kdiv = (KSZ > 0) ? KSZ : 1;   // KSZ==0 only in dead (dense) paths
  int c = kg / kdiv;
  int r = kg - c * kdiv;
  int src = mg * p.convStride + r * p.convDil - p.convPad;
  return (src >= 0 && src < p.convLin) ? Ab[(long)c * p.convLin + src] : 0.f;
}

template<int AMODE, int TRANSB, int CSTORT, int ACT, int KSZ, int TDM>
__global__ __launch_bounds__(256)
void gemm_wmma(GemmP p)
{
  constexpr bool useTdm = (TDM != 0) && (USE_TDM != 0);
  constexpr bool tdmA   = useTdm && (AMODE == 0);

  // bf16 operand buffers: double-buffered; n-major B so every lane's fragment
  // is two contiguous 16B runs -> ds_load_b128 pairs.
  __shared__ unsigned int la_u[2][TILE_M][TILE_K / 2];  // 16 KB
  __shared__ unsigned int lb_u[2][TILE_N][TILE_K / 2];  //  8 KB
  // fp32 DMA landing zones: single-buffered (consumed before next issue)
  __shared__ float la_f[TILE_M][TILE_K];                // 16 KB
  __shared__ float lb_f[TILE_K * TILE_N];               //  8 KB

  const int tid  = threadIdx.x;
  const int lane = tid & 31;
  const int wave = tid >> 5;
  const int wm = wave >> 1;   // 0..3 : 32 rows each
  const int wn = wave & 1;    // 0..1 : 32 cols each
  const int g  = lane >> 4;   // lane group 0/1
  const int lr = lane & 15;

  const int tm = blockIdx.x * TILE_M;
  const int tn = blockIdx.y * TILE_N;
  const int z  = blockIdx.z;
  const int outer = z / p.innerCount;
  const int inner = z - outer * p.innerCount;

  const float* Ab = p.A + (long)outer * p.aOut + (long)inner * p.aIn;
  const float* Bb = p.B + (long)outer * p.bOut + (long)inner * p.bIn;
  float*       Cb = p.C + (long)outer * p.cOut + (long)inner * p.cIn;
  const float* Rb = p.res ? (p.res + (long)outer * p.cOut + (long)inner * p.cIn) : nullptr;

  v8f acc[2][2] = {{{}, {}}, {{}, {}}};

  // ---- stage issue: start DMA / cooperative loads for K-tile k0 into buf ----
  auto issueStage = [&](int k0, int buf) {
    // A panel
    if (tdmA) {
      if (tid < 32)
        tdm_issue(Ab + (long)tm * p.lda + k0, &la_f[0][0],
                  p.K - k0, p.M - tm, p.lda, TILE_K, TILE_M);
    } else if (AMODE == 1) {
      #pragma unroll
      for (int t = 0; t < (TILE_M * TILE_K / 2) / 256; ++t) {
        int pp = tid + t * 256;
        int ml = pp >> 4;
        int kp = (pp & 15) * 2;
        int mg = tm + ml;
        int kg = k0 + kp;
        float v0 = 0.f, v1 = 0.f;
        if (mg < p.M) {
          if (kg < p.K)     v0 = fetchConv<KSZ>(Ab, mg, kg, p);
          if (kg + 1 < p.K) v1 = fetchConv<KSZ>(Ab, mg, kg + 1, p);
          if (t == 0 && kg + TILE_K < p.K) {
            constexpr int kdiv = (KSZ > 0) ? KSZ : 1;
            int c = (kg + TILE_K) / kdiv;
            int r = (kg + TILE_K) - c * kdiv;
            int src = mg * p.convStride + r * p.convDil - p.convPad;
            if (src >= 0 && src < p.convLin)
              __builtin_prefetch(&Ab[(long)c * p.convLin + src], 0, 1);
          }
        }
        la_u[buf][ml][kp >> 1] = pack2bf(v0, v1);
      }
    } else {
      // dense cooperative fallback (USE_TDM == 0)
      #pragma unroll
      for (int t = 0; t < (TILE_M * TILE_K / 2) / 256; ++t) {
        int pp = tid + t * 256;
        int ml = pp >> 4;
        int kp = (pp & 15) * 2;
        int mg = tm + ml;
        int kg = k0 + kp;
        float v0 = 0.f, v1 = 0.f;
        if (mg < p.M) {
          const float* ar = Ab + (long)mg * p.lda;
          if (kg < p.K)     v0 = ar[kg];
          if (kg + 1 < p.K) v1 = ar[kg + 1];
        }
        la_u[buf][ml][kp >> 1] = pack2bf(v0, v1);
      }
    }
    // B panel
    if (useTdm) {
      if (tid < 32) {
        if (TRANSB)   // rows = N, cols = K
          tdm_issue(Bb + (long)tn * p.ldb + k0, &lb_f[0],
                    p.K - k0, p.N - tn, p.ldb, TILE_K, TILE_N);
        else          // rows = K, cols = N (transposing conversion below)
          tdm_issue(Bb + (long)k0 * p.ldb + tn, &lb_f[0],
                    p.N - tn, p.K - k0, p.ldb, TILE_N, TILE_K);
      }
    } else {
      #pragma unroll
      for (int t = 0; t < (TILE_N * TILE_K / 2) / 256; ++t) {
        int pp = tid + t * 256;
        int nl, kg;
        float v0 = 0.f, v1 = 0.f;
        if (TRANSB) {
          nl = pp >> 4;
          int kp = (pp & 15) * 2;
          kg = k0 + kp;
          int ng = tn + nl;
          if (ng < p.N) {
            const float* br = Bb + (long)ng * p.ldb;
            if (kg < p.K)     v0 = br[kg];
            if (kg + 1 < p.K) v1 = br[kg + 1];
          }
          lb_u[buf][nl][kp >> 1] = pack2bf(v0, v1);
        } else {
          int kp2 = pp >> 6;
          nl = pp & 63;
          kg = k0 + kp2 * 2;
          int ng = tn + nl;
          if (ng < p.N) {
            if (kg < p.K)     v0 = Bb[(long)kg * p.ldb + ng];
            if (kg + 1 < p.K) v1 = Bb[(long)(kg + 1) * p.ldb + ng];
          }
          lb_u[buf][nl][kp2] = pack2bf(v0, v1);
        }
      }
    }
  };

  // ---- stage complete: wait DMA, convert fp32 -> packed bf16 ----
  auto completeStage = [&](int buf) {
    if (useTdm) {
      if (tid < 32) __builtin_amdgcn_s_wait_tensorcnt(0);
      __syncthreads();                       // DMA data visible to all waves
      if (tdmA) {
        #pragma unroll
        for (int t = 0; t < (TILE_M * TILE_K / 2) / 256; ++t) {
          int pp = tid + t * 256;
          int ml = pp >> 4;
          int kp = (pp & 15) * 2;
          la_u[buf][ml][kp >> 1] = pack2bf(la_f[ml][kp], la_f[ml][kp + 1]);
        }
      }
      #pragma unroll
      for (int t = 0; t < (TILE_N * TILE_K / 2) / 256; ++t) {
        int pp = tid + t * 256;
        int nl = pp >> 4;
        int kp = (pp & 15) * 2;
        float v0, v1;
        if (TRANSB) { v0 = lb_f[nl * TILE_K + kp];  v1 = lb_f[nl * TILE_K + kp + 1]; }
        else        { v0 = lb_f[kp * TILE_N + nl];  v1 = lb_f[(kp + 1) * TILE_N + nl]; }
        lb_u[buf][nl][kp >> 1] = pack2bf(v0, v1);
      }
    }
    __syncthreads();                         // staged bf16 panels visible
  };

  // ---- software pipeline: DMA/loads of tile i+1 overlap WMMA of tile i ----
  issueStage(0, 0);
  completeStage(0);
  int cur = 0;
  for (int k0 = 0; k0 < p.K; k0 += TILE_K) {
    const int nxt = cur ^ 1;
    const bool more = (k0 + TILE_K) < p.K;   // uniform across workgroup
    if (more) issueStage(k0 + TILE_K, nxt);

    // ---- fragments: two contiguous 16B LDS loads each ----
    v16bf fa[2], fb[2];
    #pragma unroll
    for (int s = 0; s < 2; ++s) {
      const v8bf* ap = (const v8bf*)&la_u[cur][wm * 32 + s * 16 + lr][0];
      fa[s] = __builtin_shufflevector(ap[g], ap[2 + g],
                0,1,2,3,4,5,6,7,8,9,10,11,12,13,14,15);
    }
    #pragma unroll
    for (int t = 0; t < 2; ++t) {
      const v8bf* bp = (const v8bf*)&lb_u[cur][wn * 32 + t * 16 + lr][0];
      fb[t] = __builtin_shufflevector(bp[g], bp[2 + g],
                0,1,2,3,4,5,6,7,8,9,10,11,12,13,14,15);
    }
    #pragma unroll
    for (int s = 0; s < 2; ++s)
      #pragma unroll
      for (int t = 0; t < 2; ++t)
        acc[s][t] = __builtin_amdgcn_wmma_f32_16x16x32_bf16(
            false, fa[s], false, fb[t], (short)0, acc[s][t], false, false);

    if (more) completeStage(nxt);
    cur = nxt;
  }

  // ---- epilogue ----
  #pragma unroll
  for (int s = 0; s < 2; ++s) {
    #pragma unroll
    for (int r = 0; r < 8; ++r) {
      int m = tm + wm * 32 + s * 16 + g * 8 + r;
      if (m >= p.M) continue;
      #pragma unroll
      for (int t = 0; t < 2; ++t) {
        int n = tn + wn * 32 + t * 16 + lr;
        if (n >= p.N) continue;
        float v = acc[s][t][r];
        if (p.bias) v += p.bias[n];
        if (ACT == 1) v = fmaxf(v, 0.f);
        if (ACT == 2) v = gelu_f(v);
        long ci = CSTORT ? ((long)n * p.ldc + m) : ((long)m * p.ldc + n);
        if (Rb) v += Rb[ci];
        Cb[ci] = v;
      }
    }
  }
}

// ---------------- small fp32 kernels ----------------

__global__ void rmsnorm_kernel(const float* x, const float* w, float* out, int D)
{
  __shared__ float red[256];
  long row = blockIdx.x;
  const float* xr = x + row * D;
  float s = 0.f;
  for (int d = threadIdx.x; d < D; d += 256) { float v = xr[d]; s += v * v; }
  red[threadIdx.x] = s; __syncthreads();
  for (int o = 128; o > 0; o >>= 1) {
    if (threadIdx.x < o) red[threadIdx.x] += red[threadIdx.x + o];
    __syncthreads();
  }
  float scale = rsqrtf(red[0] / (float)D + 1e-8f);
  for (int d = threadIdx.x; d < D; d += 256) out[row * D + d] = xr[d] * scale * w[d];
}

__global__ void rope_kernel(float* x, long total)
{
  long idx = (long)blockIdx.x * blockDim.x + threadIdx.x;
  if (idx >= total) return;
  const int half = HD / 2;
  int j = (int)(idx % half); long r = idx / half;
  int h = (int)(r % HEADS); r /= HEADS;
  int t = (int)(r % CTX);   int b = (int)(r / CTX);
  long base = ((long)(b * CTX + t)) * DIMS + (long)h * HD;
  float inv = powf(10000.f, -(float)j / (float)half);
  float fr = (float)t * inv;
  float c = cosf(fr), s = sinf(fr);
  const float sc = 0.35355339059327379f; // 64^-0.25
  float a = x[base + j], d = x[base + half + j];
  x[base + j]        = (a * c - d * s) * sc;
  x[base + half + j] = (a * s + d * c) * sc;
}

__global__ void mask_softmax_kernel(float* scores, const float* kact, const float* factor)
{
  const int q = blockIdx.x, h = blockIdx.y, b = blockIdx.z;
  const int tid = threadIdx.x;
  float f  = factor[0];
  float zf = log1pf(expf(f));
  zf = fminf(fmaxf(zf, 1e-5f), 0.1f);
  float* row = scores + (((long)b * HEADS + h) * CTX + q) * (long)CTX;
  const int NI = (CTX + 255) / 256;
  float lv[NI];
  float mx = -3.4e38f;
  #pragma unroll
  for (int i = 0; i < NI; ++i) {
    int k = tid + i * 256;
    float v = 0.f;
    if (k < CTX) {
      float s  = row[k];
      float k0 = kact[((long)b * CTX + k) * DIMS + h * HD];
      float sz = (k0 == 0.f) ? zf : 1.f;
      v = (k <= q) ? s * sz : 0.f;
      mx = fmaxf(mx, v);
    }
    lv[i] = v;
  }
  __shared__ float red[256];
  red[tid] = mx; __syncthreads();
  for (int o = 128; o > 0; o >>= 1) {
    if (tid < o) red[tid] = fmaxf(red[tid], red[tid + o]);
    __syncthreads();
  }
  mx = red[0]; __syncthreads();
  float sum = 0.f;
  #pragma unroll
  for (int i = 0; i < NI; ++i) {
    int k = tid + i * 256;
    if (k < CTX) { float e = expf(lv[i] - mx); lv[i] = e; sum += e; }
  }
  red[tid] = sum; __syncthreads();
  for (int o = 128; o > 0; o >>= 1) {
    if (tid < o) red[tid] += red[tid + o];
    __syncthreads();
  }
  float inv = 1.f / red[0];
  #pragma unroll
  for (int i = 0; i < NI; ++i) {
    int k = tid + i * 256;
    if (k < CTX) row[k] = lv[i] * inv;
  }
}

__global__ void dwconv3_kernel(const float* in, const float* w, const float* bias,
                               float* out, long total)
{
  long idx = (long)blockIdx.x * blockDim.x + threadIdx.x;
  if (idx >= total) return;
  int t = (int)(idx % CTX);
  long bc = idx / CTX;
  int c = (int)(bc % DIMS);
  float s = bias[c];
  #pragma unroll
  for (int r = 0; r < 3; ++r) {
    int tt = t + r - 1;
    if (tt >= 0 && tt < CTX) s += in[bc * CTX + tt] * w[c * 3 + r];
  }
  out[idx] = s;
}

__global__ void weconv1_kernel(const float* w, const float* wt, const float* bias,
                               float* out, long total)
{
  long idx = (long)blockIdx.x * blockDim.x + threadIdx.x;
  if (idx >= total) return;
  const int Lout = 30000;
  int t = (int)(idx % Lout);
  long bc = idx / Lout;
  int c = (int)(bc % DIMS);
  int b = (int)(bc / DIMS);
  float s = bias[c];
  #pragma unroll
  for (int r = 0; r < 11; ++r) {
    int src = 5 * t + r - 5;
    if (src >= 0 && src < NSAMP) s += w[(long)b * NSAMP + src] * wt[c * 11 + r];
  }
  out[idx] = gelu_f(s);
}

__global__ void mean_kernel(const float* h, float* y, int L)
{
  __shared__ float red[256];
  long bc = blockIdx.x;
  const float* p = h + bc * L;
  float s = 0.f;
  for (int t = threadIdx.x; t < L; t += 256) s += p[t];
  red[threadIdx.x] = s; __syncthreads();
  for (int o = 128; o > 0; o >>= 1) {
    if (threadIdx.x < o) red[threadIdx.x] += red[threadIdx.x + o];
    __syncthreads();
  }
  if (threadIdx.x == 0) y[bc] = red[0] / (float)L;
}

__global__ void fc_kernel(const float* y, const float* w, const float* b, float* out,
                          int Bn, int O, int I, int act)
{
  int idx = blockIdx.x * blockDim.x + threadIdx.x;
  if (idx >= Bn * O) return;
  int o = idx % O, bb = idx / O;
  float s = b[o];
  for (int k = 0; k < I; ++k) s += y[bb * I + k] * w[o * I + k];
  out[idx] = (act == 1) ? fmaxf(s, 0.f) : 1.f / (1.f + expf(-s));
}

__global__ void scale_gelu_kernel(float* h, const float* y, long total)
{
  long idx = (long)blockIdx.x * blockDim.x + threadIdx.x;
  if (idx >= total) return;
  long bc = idx / CTX; // h layout [b, c, t] ; y layout [b, c]
  h[idx] = gelu_f(h[idx] * y[bc]);
}

__global__ void pool10_kernel(const float* g, float* o, long total)
{
  long idx = (long)blockIdx.x * blockDim.x + threadIdx.x;
  if (idx >= total) return;
  int t = (int)(idx % CTX);
  long bc = idx / CTX;
  const float* p = g + bc * 15000 + (long)t * 10;
  float s = 0.f;
  #pragma unroll
  for (int i = 0; i < 10; ++i) s += p[i];
  o[idx] = s * 0.1f;
}

__global__ void blend_kernel(const float* spec, const float* wav, const float* bsw,
                             float* h, long total)
{
  long idx = (long)blockIdx.x * blockDim.x + threadIdx.x;
  if (idx >= total) return;
  int d = (int)(idx % DIMS);
  long r = idx / DIMS;
  int t = (int)(r % CTX);
  int b = (int)(r / CTX);
  float bl = 1.f / (1.f + expf(-bsw[0]));
  long src = ((long)b * DIMS + d) * CTX + t;
  float sp = spec[src];
  float wv = wav[src];
  const int half = DIMS / 2;
  const float inc = logf(10000.f) / (float)(half - 1);
  int jj = (d < half) ? d : d - half;
  float invf = expf(-inc * (float)jj);
  float ang  = (float)t * invf;
  float pos  = (d < half) ? sinf(ang) : cosf(ang);
  h[idx] = bl * sp + (1.f - bl) * wv + pos;
}

__global__ void add_kernel(const float* a, const float* b, float* out, long total)
{
  long idx = (long)blockIdx.x * blockDim.x + threadIdx.x;
  if (idx >= total) return;
  out[idx] = a[idx] + b[idx];
}

// ---------------- host-side helpers ----------------
static inline int ceilDiv(int a, int b) { return (a + b - 1) / b; }

static GemmP mkP(const float* A, const float* B, const float* bias, const float* res, float* C,
                 int M, int N, int K, int lda, int ldb, int ldc, int innerCount,
                 long aIn, long aOut, long bIn, long bOut, long cIn, long cOut,
                 int cLin, int cStr, int cPad, int cK, int cDil)
{
  GemmP p;
  p.A = A; p.B = B; p.bias = bias; p.res = res; p.C = C;
  p.M = M; p.N = N; p.K = K; p.lda = lda; p.ldb = ldb; p.ldc = ldc;
  p.innerCount = innerCount;
  p.aIn = aIn; p.aOut = aOut; p.bIn = bIn; p.bOut = bOut; p.cIn = cIn; p.cOut = cOut;
  p.convLin = cLin; p.convStride = cStr; p.convPad = cPad; p.convK = cK; p.convDil = cDil;
  return p;
}

extern "C" void kernel_launch(void* const* d_in, const int* in_sizes, int n_in,
                              void* d_out, int out_size, void* d_ws, size_t ws_size,
                              hipStream_t stream)
{
  // ---- inputs (setup_inputs order) ----
  const float* x       = (const float*)d_in[0];
  const float* wav     = (const float*)d_in[1];
  const float* se_w1   = (const float*)d_in[2];
  const float* se_b1   = (const float*)d_in[3];
  const float* se_w2   = (const float*)d_in[4];
  const float* se_b2   = (const float*)d_in[5];
  const float* se_w3   = (const float*)d_in[6];
  const float* se_b3   = (const float*)d_in[7];
  const float* se_w4   = (const float*)d_in[8];
  const float* se_b4   = (const float*)d_in[9];
  const float* se_fc1w = (const float*)d_in[10];
  const float* se_fc1b = (const float*)d_in[11];
  const float* se_fc2w = (const float*)d_in[12];
  const float* se_fc2b = (const float*)d_in[13];
  const float* se_w5   = (const float*)d_in[14];
  const float* se_b5   = (const float*)d_in[15];
  const float* we_w1   = (const float*)d_in[16];
  const float* we_b1   = (const float*)d_in[17];
  const float* we_w2   = (const float*)d_in[18];
  const float* we_b2   = (const float*)d_in[19];
  const float* qw      = (const float*)d_in[20];
  const float* qb      = (const float*)d_in[21];
  const float* kw      = (const float*)d_in[22];
  const float* vw      = (const float*)d_in[23];
  const float* vb      = (const float*)d_in[24];
  const float* ow      = (const float*)d_in[25];
  const float* ob      = (const float*)d_in[26];
  const float* factor  = (const float*)d_in[27];
  const float* lna_w   = (const float*)d_in[28];
  const float* lnc_w   = (const float*)d_in[29];
  const float* m1w     = (const float*)d_in[30];
  const float* m1b     = (const float*)d_in[31];
  const float* m2w     = (const float*)d_in[32];
  const float* m2b     = (const float*)d_in[33];
  const float* ln_enc  = (const float*)d_in[34];
  const float* bsw     = (const float*)d_in[35];

  float* outp = (float*)d_out;

  // ---- workspace bump allocation (floats) ----
  const size_t S = (size_t)BATCH * DIMS * CTX; // 3,072,000
  float* ws   = (float*)d_ws;
  float* s1   = ws;            ws += S;
  float* s2   = ws;            ws += S;
  float* yv   = ws;            ws += BATCH * DIMS;
  float* y2v  = ws;            ws += BATCH * (DIMS / 16);
  float* y3v  = ws;            ws += BATCH * DIMS;
  float* wpool= ws;            ws += S;
  float* hbuf = ws;            ws += S;
  float* ubuf = ws;            ws += S;
  float* qbuf = ws;            ws += S;
  float* kbuf = ws;            ws += S;
  float* vbuf = ws;            ws += S;
  float* h1buf= ws;            ws += S;
  float* aobuf= ws;            ws += S;
  float* ffbuf= ws;            ws += (size_t)BATCH * CTX * FF;
  float* big  = ws;            // union: g1(61.44M)+g2(30.72M)  vs  scores(72M)
  float* g1   = big;
  float* g2   = big + (size_t)BATCH * DIMS * 30000;
  float* scores = big;

  const dim3 blk(256);
  const long SEL = (long)BATCH * DIMS * CTX;

  // ================= spectrogram stem =================
  {
    // se1: conv(128->1024, k3 p1) + gelu  [implicit-GEMM, WMMA, TDM weights]
    GemmP p = mkP(x, se_w1, se_b1, nullptr, s1,
                  CTX, DIMS, MELS * 3, 0, MELS * 3, CTX, 1,
                  0, (long)MELS * CTX, 0, 0, 0, (long)DIMS * CTX,
                  CTX, 1, 1, 3, 1);
    gemm_wmma<1,1,1,2,3,1><<<dim3(ceilDiv(CTX,TILE_M), ceilDiv(DIMS,TILE_N), BATCH), blk, 0, stream>>>(p);
  }
  {
    // se2: conv(1024->1024, k3 p2 d2)
    GemmP p = mkP(s1, se_w2, se_b2, nullptr, s2,
                  CTX, DIMS, DIMS * 3, 0, DIMS * 3, CTX, 1,
                  0, (long)DIMS * CTX, 0, 0, 0, (long)DIMS * CTX,
                  CTX, 1, 2, 3, 2);
    gemm_wmma<1,1,1,0,3,1><<<dim3(ceilDiv(CTX,TILE_M), ceilDiv(DIMS,TILE_N), BATCH), blk, 0, stream>>>(p);
  }
  // se3: depthwise conv k3 p1
  dwconv3_kernel<<<(SEL + 255) / 256, blk, 0, stream>>>(s2, se_w3, se_b3, s1, SEL);
  {
    // se4: pointwise conv (k1)
    GemmP p = mkP(s1, se_w4, se_b4, nullptr, s2,
                  CTX, DIMS, DIMS, 0, DIMS, CTX, 1,
                  0, (long)DIMS * CTX, 0, 0, 0, (long)DIMS * CTX,
                  CTX, 1, 0, 1, 1);
    gemm_wmma<1,1,1,0,1,1><<<dim3(ceilDiv(CTX,TILE_M), ceilDiv(DIMS,TILE_N), BATCH), blk, 0, stream>>>(p);
  }
  // SE block
  mean_kernel<<<BATCH * DIMS, blk, 0, stream>>>(s2, yv, CTX);
  fc_kernel<<<1, blk, 0, stream>>>(yv, se_fc1w, se_fc1b, y2v, BATCH, DIMS/16, DIMS, 1);
  fc_kernel<<<ceilDiv(BATCH*DIMS,256), blk, 0, stream>>>(y2v, se_fc2w, se_fc2b, y3v, BATCH, DIMS, DIMS/16, 2);
  scale_gelu_kernel<<<(SEL + 255) / 256, blk, 0, stream>>>(s2, y3v, SEL);
  {
    // se5: conv(1024->1024, k3 p1)
    GemmP p = mkP(s2, se_w5, se_b5, nullptr, s1,
                  CTX, DIMS, DIMS * 3, 0, DIMS * 3, CTX, 1,
                  0, (long)DIMS * CTX, 0, 0, 0, (long)DIMS * CTX,
                  CTX, 1, 1, 3, 1);
    gemm_wmma<1,1,1,0,3,1><<<dim3(ceilDiv(CTX,TILE_M), ceilDiv(DIMS,TILE_N), BATCH), blk, 0, stream>>>(p);
  }

  // ================= waveform stem =================
  {
    long tot = (long)BATCH * DIMS * 30000;
    weconv1_kernel<<<(tot + 255) / 256, blk, 0, stream>>>(wav, we_w1, we_b1, g1, tot);
  }
  {
    // we2: conv(1024->1024, k5 s2 p2) + gelu — dominant FLOPs, implicit-GEMM on WMMA
    GemmP p = mkP(g1, we_w2, we_b2, nullptr, g2,
                  15000, DIMS, DIMS * 5, 0, DIMS * 5, 15000, 1,
                  0, (long)DIMS * 30000, 0, 0, 0, (long)DIMS * 15000,
                  30000, 2, 2, 5, 1);
    gemm_wmma<1,1,1,2,5,1><<<dim3(ceilDiv(15000,TILE_M), ceilDiv(DIMS,TILE_N), BATCH), blk, 0, stream>>>(p);
  }
  pool10_kernel<<<(SEL + 255) / 256, blk, 0, stream>>>(g2, wpool, SEL);

  // blend + sinusoidal positions -> hbuf [B, CTX, DIMS]
  blend_kernel<<<(SEL + 255) / 256, blk, 0, stream>>>(s1, wpool, bsw, hbuf, SEL);

  // ================= transformer layers =================
  const int ROWS = BATCH * CTX;
  const long ropeTot = (long)BATCH * CTX * HEADS * (HD / 2);
  for (int i = 0; i < LAYERS; ++i) {
    const float* qwi = qw + (size_t)i * DIMS * DIMS;
    const float* kwi = kw + (size_t)i * DIMS * DIMS;
    const float* vwi = vw + (size_t)i * DIMS * DIMS;
    const float* owi = ow + (size_t)i * DIMS * DIMS;
    const float* m1wi = m1w + (size_t)i * FF * DIMS;
    const float* m2wi = m2w + (size_t)i * DIMS * FF;

    // u = rms_norm(h, lna)
    rmsnorm_kernel<<<ROWS, blk, 0, stream>>>(hbuf, lna_w + (size_t)i * DIMS, ubuf, DIMS);

    // q / k / v projections
    {
      GemmP p = mkP(ubuf, qwi, qb + (size_t)i * DIMS, nullptr, qbuf,
                    ROWS, DIMS, DIMS, DIMS, DIMS, DIMS, 1, 0,0,0,0,0,0, 0,0,0,1,1);
      gemm_wmma<0,1,0,0,0,1><<<dim3(ceilDiv(ROWS,TILE_M), ceilDiv(DIMS,TILE_N), 1), blk, 0, stream>>>(p);
    }
    {
      GemmP p = mkP(ubuf, kwi, nullptr, nullptr, kbuf,
                    ROWS, DIMS, DIMS, DIMS, DIMS, DIMS, 1, 0,0,0,0,0,0, 0,0,0,1,1);
      gemm_wmma<0,1,0,0,0,1><<<dim3(ceilDiv(ROWS,TILE_M), ceilDiv(DIMS,TILE_N), 1), blk, 0, stream>>>(p);
    }
    {
      GemmP p = mkP(ubuf, vwi, vb + (size_t)i * DIMS, nullptr, vbuf,
                    ROWS, DIMS, DIMS, DIMS, DIMS, DIMS, 1, 0,0,0,0,0,0, 0,0,0,1,1);
      gemm_wmma<0,1,0,0,0,1><<<dim3(ceilDiv(ROWS,TILE_M), ceilDiv(DIMS,TILE_N), 1), blk, 0, stream>>>(p);
    }

    // RoPE (+ HD^-0.25 scale) on q and k
    rope_kernel<<<(ropeTot + 255) / 256, blk, 0, stream>>>(qbuf, ropeTot);
    rope_kernel<<<(ropeTot + 255) / 256, blk, 0, stream>>>(kbuf, ropeTot);

    // scores = q @ k^T per (b, h)
    {
      GemmP p = mkP(qbuf, kbuf, nullptr, nullptr, scores,
                    CTX, CTX, HD, DIMS, DIMS, CTX, HEADS,
                    HD, (long)CTX * DIMS, HD, (long)CTX * DIMS,
                    (long)CTX * CTX, (long)HEADS * CTX * CTX,
                    0,0,0,1,1);
      gemm_wmma<0,1,0,0,0,1><<<dim3(ceilDiv(CTX,TILE_M), ceilDiv(CTX,TILE_N), BATCH * HEADS), blk, 0, stream>>>(p);
    }

    // causal mask * token scale, softmax
    mask_softmax_kernel<<<dim3(CTX, HEADS, BATCH), blk, 0, stream>>>(scores, kbuf, factor + i);

    // o = att @ v  -> head-interleaved [B, CTX, DIMS]  (B panel K-major via TDM)
    {
      GemmP p = mkP(scores, vbuf, nullptr, nullptr, aobuf,
                    CTX, HD, CTX, CTX, DIMS, DIMS, HEADS,
                    (long)CTX * CTX, (long)HEADS * CTX * CTX,
                    HD, (long)CTX * DIMS, HD, (long)CTX * DIMS,
                    0,0,0,1,1);
      gemm_wmma<0,0,0,0,0,1><<<dim3(ceilDiv(CTX,TILE_M), 1, BATCH * HEADS), blk, 0, stream>>>(p);
    }

    // h1 = h + o @ ow^T + ob
    {
      GemmP p = mkP(aobuf, owi, ob + (size_t)i * DIMS, hbuf, h1buf,
                    ROWS, DIMS, DIMS, DIMS, DIMS, DIMS, 1, 0,0,0,0,0,0, 0,0,0,1,1);
      gemm_wmma<0,1,0,0,0,1><<<dim3(ceilDiv(ROWS,TILE_M), ceilDiv(DIMS,TILE_N), 1), blk, 0, stream>>>(p);
    }

    // m = rms_norm(h1, lnc)
    rmsnorm_kernel<<<ROWS, blk, 0, stream>>>(h1buf, lnc_w + (size_t)i * DIMS, ubuf, DIMS);

    // ff = relu(m @ m1w^T + m1b)
    {
      GemmP p = mkP(ubuf, m1wi, m1b + (size_t)i * FF, nullptr, ffbuf,
                    ROWS, FF, DIMS, DIMS, DIMS, FF, 1, 0,0,0,0,0,0, 0,0,0,1,1);
      gemm_wmma<0,1,0,1,0,1><<<dim3(ceilDiv(ROWS,TILE_M), ceilDiv(FF,TILE_N), 1), blk, 0, stream>>>(p);
    }
    // h2 = h1 + ff @ m2w^T + m2b   (reuse qbuf as h2)
    {
      GemmP p = mkP(ffbuf, m2wi, m2b + (size_t)i * DIMS, h1buf, qbuf,
                    ROWS, DIMS, FF, FF, FF, DIMS, 1, 0,0,0,0,0,0, 0,0,0,1,1);
      gemm_wmma<0,1,0,0,0,1><<<dim3(ceilDiv(ROWS,TILE_M), ceilDiv(DIMS,TILE_N), 1), blk, 0, stream>>>(p);
    }
    // h = h2 + r  (r = layer input, still in hbuf)
    add_kernel<<<(SEL + 255) / 256, blk, 0, stream>>>(qbuf, hbuf, hbuf, SEL);
  }

  // final rms_norm -> output
  rmsnorm_kernel<<<ROWS, blk, 0, stream>>>(hbuf, ln_enc, outp, DIMS);
}
